// GTMKT_40261023433345
// MI455X (gfx1250) — matrix operand order repaired and checked by reading, not compiled, not run on hardware
//
#include <hip/hip_runtime.h>
#include <hip/hip_bf16.h>

#define NN   32768
#define FF   128
#define DD   128
#define EE   524288
#define LL   8
#define HH   8
#define CSZ  32
#define NCLS 10
#define KC   (NN/CSZ)   /* 1024 clusters */
#define TT   (KC+1)     /* 1025 tokens   */
#define ETOT (EE+NN)    /* edges + self loops */
#define DH   (DD/HH)    /* 16 head dim   */

typedef __attribute__((ext_vector_type(16))) __bf16 v16bf;
typedef __attribute__((ext_vector_type(8)))  float  v8f;

__device__ __forceinline__ float gelu_tanh(float x) {
    float x3 = x * x * x;
    return 0.5f * x * (1.0f + tanhf(0.7978845608028654f * (x + 0.044715f * x3)));
}

__device__ __forceinline__ void atomicMaxF(float* addr, float v) {
    unsigned int* ua = (unsigned int*)addr;
    unsigned int old = *ua;
    while (__uint_as_float(old) < v) {
        unsigned int assumed = old;
        old = atomicCAS(ua, assumed, __float_as_uint(v));
        if (old == assumed) break;
    }
}

// ---------------------------------------------------------------------------
// Weight pre-pass: W[K,N] f32 (row major)  ->  Wt[N,K] bf16 (transposed)
// Makes every B fragment two contiguous, 32B-aligned b128 loads per lane.
// ---------------------------------------------------------------------------
__global__ void convert_wT(const float* __restrict__ W, __bf16* __restrict__ Wt,
                           int K, int N) {
    int i = blockIdx.x * blockDim.x + threadIdx.x;
    if (i >= K * N) return;
    int k = i / N, n = i % N;
    Wt[(size_t)n * K + k] = (__bf16)W[i];
}

// ---------------------------------------------------------------------------
// GEMM: C[M,N] = act(A[M,K] @ B[K,N] + bias) (+= if accum), B given as
// transposed bf16 Bt[N,K]. Each wave32 owns a 16x64 C tile: one A fragment
// is reused by 4 v_wmma_f32_16x16x32_bf16 per K-step of 32.
//   A frag (16x32 bf16): lane half h, elem e   -> K=h*16+e, row=lane&15
//   B frag (32x16 bf16): lane half h, elem e   -> K=h*16+e, col=lane&15
//     -> in Bt[N,K] layout that is 16 *contiguous* bf16 at Bt[col*K + k0 + h*16]
//   C/D (8 VGPR f32): c[r] -> row r+8*h, col lane&15
// ---------------------------------------------------------------------------
__global__ void gemm_wmma_bf16(const float*  __restrict__ A,
                               const __bf16* __restrict__ Bt,
                               const float*  __restrict__ bias,
                               float*        __restrict__ C,
                               int M, int Kdim, int Ndim, int act, int accum) {
    int wave    = blockIdx.x * (blockDim.x >> 5) + (threadIdx.x >> 5);
    int tilesN  = Ndim >> 6;                       // 64-wide column tiles
    int tilesM  = (M + 15) >> 4;
    if (wave >= tilesM * tilesN) return;
    int tm   = (wave / tilesN) << 4;
    int tn   = (wave % tilesN) << 6;
    int lane = threadIdx.x & 31;
    int half = lane >> 4;
    int l16  = lane & 15;

    int arow = tm + l16;
    if (arow > M - 1) arow = M - 1;                // clamp ragged-M loads
    const float* Abase = A + (size_t)arow * Kdim;

    // 4 contiguous 16-col B panels; each lane reads 16 contiguous bf16 per panel
    const __bf16* Bp[4];
#pragma unroll
    for (int c = 0; c < 4; ++c)
        Bp[c] = Bt + (size_t)(tn + c * 16 + l16) * Kdim + half * 16;

    v8f acc0 = {0.f,0.f,0.f,0.f,0.f,0.f,0.f,0.f};
    v8f acc1 = acc0, acc2 = acc0, acc3 = acc0;

    for (int k0 = 0; k0 < Kdim; k0 += 32) {
        const float* ap = Abase + k0 + half * 16;
        __builtin_prefetch(Abase + k0 + 32, 0, 1);     // next A slice -> global_prefetch
        const float4* apv = (const float4*)ap;         // 16B aligned (K % 32 == 0)
        float4 a0 = apv[0], a1 = apv[1];               // K = h*16  .. h*16+7
        float4 a2 = apv[2], a3 = apv[3];               // K = h*16+8.. h*16+15
        v16bf af;
        af[0]=(__bf16)a0.x; af[1]=(__bf16)a0.y; af[2]=(__bf16)a0.z; af[3]=(__bf16)a0.w;
        af[4]=(__bf16)a1.x; af[5]=(__bf16)a1.y; af[6]=(__bf16)a1.z; af[7]=(__bf16)a1.w;
        af[8]=(__bf16)a2.x; af[9]=(__bf16)a2.y; af[10]=(__bf16)a2.z; af[11]=(__bf16)a2.w;
        af[12]=(__bf16)a3.x; af[13]=(__bf16)a3.y; af[14]=(__bf16)a3.z; af[15]=(__bf16)a3.w;

        v16bf b0 = *(const v16bf*)(Bp[0] + k0);
        v16bf b1 = *(const v16bf*)(Bp[1] + k0);
        v16bf b2 = *(const v16bf*)(Bp[2] + k0);
        v16bf b3 = *(const v16bf*)(Bp[3] + k0);

        acc0 = __builtin_amdgcn_wmma_f32_16x16x32_bf16(false, af, false, b0, (short)0, acc0, false, false);
        acc1 = __builtin_amdgcn_wmma_f32_16x16x32_bf16(false, af, false, b1, (short)0, acc1, false, false);
        acc2 = __builtin_amdgcn_wmma_f32_16x16x32_bf16(false, af, false, b2, (short)0, acc2, false, false);
        acc3 = __builtin_amdgcn_wmma_f32_16x16x32_bf16(false, af, false, b3, (short)0, acc3, false, false);
    }

    v8f accs[4] = {acc0, acc1, acc2, acc3};
#pragma unroll
    for (int c = 0; c < 4; ++c) {
        int col = tn + c * 16 + l16;
#pragma unroll
        for (int r = 0; r < 8; ++r) {
            int row = tm + r + half * 8;
            if (row < M) {
                float v = accs[c][r];
                if (bias) v += bias[col];
                if (act == 1) v = gelu_tanh(v);
                float* o = C + (size_t)row * Ndim + col;
                if (accum) *o += v; else *o = v;
            }
        }
    }
}

// ---------------------------------------------------------------------------
// GAT pieces
// ---------------------------------------------------------------------------
__global__ void gat_scores(const float* __restrict__ g,
                           const float* __restrict__ a_s,
                           const float* __restrict__ a_d,
                           float* __restrict__ s_src, float* __restrict__ s_dst) {
    int n = blockIdx.x * blockDim.x + threadIdx.x;
    if (n >= NN) return;
    const float* r = g + (size_t)n * DD;
    float ss = 0.f, sd = 0.f;
#pragma unroll 4
    for (int d = 0; d < DD; ++d) { float h = r[d]; ss += h * a_s[d]; sd += h * a_d[d]; }
    s_src[n] = ss; s_dst[n] = sd;
}

__global__ void gat_init(float* __restrict__ m, float* __restrict__ denom,
                         float* __restrict__ outh, const float* __restrict__ bvec) {
    int i = blockIdx.x * blockDim.x + threadIdx.x;
    if (i < NN) { m[i] = -__builtin_inff(); denom[i] = 0.f; }
    if (i < NN * DD) outh[i] = bvec[i & (DD - 1)];
}

__global__ void gat_edge1(const int* __restrict__ ei,
                          const float* __restrict__ s_src, const float* __restrict__ s_dst,
                          float* __restrict__ eV, float* __restrict__ m) {
    int e = blockIdx.x * blockDim.x + threadIdx.x;
    if (e >= ETOT) return;
    int s = (e < EE) ? ei[e]      : (e - EE);
    int d = (e < EE) ? ei[EE + e] : (e - EE);
    float v = s_src[s] + s_dst[d];
    v = (v >= 0.f) ? v : 0.2f * v;          // leaky relu, slope 0.2
    eV[e] = v;
    atomicMaxF(&m[d], v);
}

__global__ void gat_edge2(const int* __restrict__ ei, float* __restrict__ eV,
                          const float* __restrict__ m, float* __restrict__ denom) {
    int e = blockIdx.x * blockDim.x + threadIdx.x;
    if (e >= ETOT) return;
    int d = (e < EE) ? ei[EE + e] : (e - EE);
    float ex = __expf(eV[e] - m[d]);
    eV[e] = ex;
    atomicAdd(&denom[d], ex);
}

__global__ void gat_edge3(const int* __restrict__ ei, const float* __restrict__ eV,
                          const float* __restrict__ denom,
                          const float* __restrict__ g, float* __restrict__ outh) {
    int e = blockIdx.x * (blockDim.x >> 5) + (threadIdx.x >> 5);   // one wave per edge
    if (e >= ETOT) return;
    int s = (e < EE) ? ei[e]      : (e - EE);
    int d = (e < EE) ? ei[EE + e] : (e - EE);
    float alpha = eV[e] / denom[d];
    int lane = threadIdx.x & 31;
    const float* gs = g + (size_t)s * DD;
    float* od = outh + (size_t)d * DD;
#pragma unroll
    for (int t = 0; t < 4; ++t) {
        int dim = lane + 32 * t;
        atomicAdd(&od[dim], gs[dim] * alpha);
    }
}

// ---------------------------------------------------------------------------
// Cluster gather + misc
// ---------------------------------------------------------------------------
__global__ void gather_nodes(const float* __restrict__ h, const int* __restrict__ order,
                             float* __restrict__ flat) {
    int i = blockIdx.x * blockDim.x + threadIdx.x;
    if (i >= NN * DD) return;
    int slot = i >> 7;
    int d    = i & 127;
    flat[i] = h[(size_t)order[slot] * DD + d];
}

__global__ void copy_cls(const float* __restrict__ cls, float* __restrict__ t) {
    int d = threadIdx.x;
    if (d < DD) t[d] = cls[d];
}

__global__ void layernorm(const float* __restrict__ x, const float* __restrict__ s,
                          const float* __restrict__ b, float* __restrict__ y, int T) {
    int q = blockIdx.x * blockDim.x + threadIdx.x;
    if (q >= T) return;
    const float* r = x + (size_t)q * DD;
    float mean = 0.f;
#pragma unroll 4
    for (int d = 0; d < DD; ++d) mean += r[d];
    mean *= (1.0f / DD);
    float var = 0.f;
#pragma unroll 4
    for (int d = 0; d < DD; ++d) { float df = r[d] - mean; var += df * df; }
    var *= (1.0f / DD);
    float inv = rsqrtf(var + 1e-5f);
    float* o = y + (size_t)q * DD;
#pragma unroll 4
    for (int d = 0; d < DD; ++d) o[d] = (r[d] - mean) * inv * s[d] + b[d];
}

// Flash-style attention: thread = (query, head), online softmax over 1025 keys.
__global__ void attention(const float* __restrict__ qkv, float* __restrict__ attno) {
    int idx = blockIdx.x * blockDim.x + threadIdx.x;
    if (idx >= TT * HH) return;
    int q = idx / HH, h = idx % HH;
    const float* Q = qkv + (size_t)q * (3 * DD) + h * DH;
    float mmax = -__builtin_inff(), ssum = 0.f;
    float acc[DH];
#pragma unroll
    for (int d = 0; d < DH; ++d) acc[d] = 0.f;
    for (int key = 0; key < TT; ++key) {
        const float* Kp = qkv + (size_t)key * (3 * DD) + DD + h * DH;
        const float* Vp = Kp + DD;
        float dot = 0.f;
#pragma unroll
        for (int d = 0; d < DH; ++d) dot += Q[d] * Kp[d];
        dot *= 0.25f;                              // dh^-0.5 = 1/4
        float nm   = fmaxf(mmax, dot);
        float corr = __expf(mmax - nm);
        float pe   = __expf(dot - nm);
        ssum = ssum * corr + pe;
#pragma unroll
        for (int d = 0; d < DH; ++d) acc[d] = acc[d] * corr + pe * Vp[d];
        mmax = nm;
    }
    float inv = 1.0f / ssum;
    float* o = attno + (size_t)q * DD + h * DH;
#pragma unroll
    for (int d = 0; d < DH; ++d) o[d] = acc[d] * inv;
}

__global__ void classifier(const float* __restrict__ t, const float* __restrict__ Wc,
                           const float* __restrict__ bc, float* __restrict__ out) {
    int c = threadIdx.x;
    if (c >= NCLS) return;
    float acc = bc[c];
#pragma unroll 4
    for (int d = 0; d < DD; ++d) acc += t[d] * Wc[d * NCLS + c];
    out[c] = acc;
}

// ---------------------------------------------------------------------------
extern "C" void kernel_launch(void* const* d_in, const int* in_sizes, int n_in,
                              void* d_out, int out_size, void* d_ws, size_t ws_size,
                              hipStream_t stream) {
    const float* x      = (const float*)d_in[0];
    const int*   ei     = (const int*)  d_in[1];
    const int*   order  = (const int*)  d_in[2];
    const float* W_in   = (const float*)d_in[3];
    const float* b_in   = (const float*)d_in[4];
    const float* Wg1    = (const float*)d_in[5];
    const float* as1    = (const float*)d_in[6];
    const float* ad1    = (const float*)d_in[7];
    const float* bg1    = (const float*)d_in[8];
    const float* Wg2    = (const float*)d_in[9];
    const float* as2    = (const float*)d_in[10];
    const float* ad2    = (const float*)d_in[11];
    const float* bg2    = (const float*)d_in[12];
    const float* W_emb  = (const float*)d_in[13];
    const float* b_emb  = (const float*)d_in[14];
    const float* clstok = (const float*)d_in[15];
    const float* ln1_s  = (const float*)d_in[16];
    const float* ln1_b  = (const float*)d_in[17];
    const float* Wqkv   = (const float*)d_in[18];
    const float* bqkv   = (const float*)d_in[19];
    const float* Wo     = (const float*)d_in[20];
    const float* bo     = (const float*)d_in[21];
    const float* ln2_s  = (const float*)d_in[22];
    const float* ln2_b  = (const float*)d_in[23];
    const float* Wf1    = (const float*)d_in[24];
    const float* bf1    = (const float*)d_in[25];
    const float* Wf2    = (const float*)d_in[26];
    const float* bf2    = (const float*)d_in[27];
    const float* Wc     = (const float*)d_in[28];
    const float* bc     = (const float*)d_in[29];
    float* out = (float*)d_out;

    // ---- workspace layout ----
    float* W = (float*)d_ws;
    size_t o = 0;
    float* hA   = W + o; o += (size_t)NN * DD;     // node buffer A
    float* hB   = W + o; o += (size_t)NN * DD;     // node buffer B
    float* G    = W + o; o += (size_t)NN * DD;     // h@Wg temp, reused as flat [K, CS*D]
    float* sS   = W + o; o += NN;
    float* sD   = W + o; o += NN;
    float* mBuf = W + o; o += NN;
    float* dBuf = W + o; o += NN;
    float* eV   = W + o; o += ETOT;                // e, then exp(e-m) in place
    float* tBuf = W + o; o += (size_t)TT * DD;
    float* uBuf = W + o; o += (size_t)TT * DD;
    float* qkvB = W + o; o += (size_t)TT * 3 * DD;
    float* atnB = W + o; o += (size_t)TT * DD;
    float* midB = W + o; o += (size_t)TT * 4 * DD;
    o = (o + 15) & ~(size_t)15;                    // 64B-align the bf16 region
    __bf16* bfbase = (__bf16*)(W + o);
    size_t bo2 = 0;
    __bf16* wtIn  = bfbase + bo2; bo2 += (size_t)FF * DD;
    __bf16* wtG1  = bfbase + bo2; bo2 += (size_t)DD * DD;
    __bf16* wtG2  = bfbase + bo2; bo2 += (size_t)DD * DD;
    __bf16* wtEmb = bfbase + bo2; bo2 += (size_t)CSZ * DD * DD;
    __bf16* wtQkv = bfbase + bo2; bo2 += (size_t)LL * DD * 3 * DD;
    __bf16* wtO   = bfbase + bo2; bo2 += (size_t)LL * DD * DD;
    __bf16* wtF1  = bfbase + bo2; bo2 += (size_t)LL * DD * 4 * DD;
    __bf16* wtF2  = bfbase + bo2; bo2 += (size_t)LL * 4 * DD * DD;

    auto conv = [&](const float* src, __bf16* dst, int K, int N) {
        int n = K * N;
        convert_wT<<<(n + 255) / 256, 256, 0, stream>>>(src, dst, K, N);
    };
    // ---- weight pre-conversion (f32 -> transposed bf16), once per call ----
    conv(W_in, wtIn, FF, DD);
    conv(Wg1, wtG1, DD, DD);
    conv(Wg2, wtG2, DD, DD);
    conv(W_emb, wtEmb, CSZ * DD, DD);
    for (int l = 0; l < LL; ++l) {
        conv(Wqkv + (size_t)l * DD * 3 * DD, wtQkv + (size_t)l * DD * 3 * DD, DD, 3 * DD);
        conv(Wo   + (size_t)l * DD * DD,     wtO   + (size_t)l * DD * DD,     DD, DD);
        conv(Wf1  + (size_t)l * DD * 4 * DD, wtF1  + (size_t)l * DD * 4 * DD, DD, 4 * DD);
        conv(Wf2  + (size_t)l * 4 * DD * DD, wtF2  + (size_t)l * 4 * DD * DD, 4 * DD, DD);
    }

    auto gemm = [&](const float* A, const __bf16* Bt, const float* bias, float* C,
                    int M, int Kd, int Nd, int act, int accum) {
        int waves  = ((M + 15) / 16) * (Nd / 64);
        int blocks = (waves + 7) / 8;
        gemm_wmma_bf16<<<blocks, 256, 0, stream>>>(A, Bt, bias, C, M, Kd, Nd, act, accum);
    };

    // 1) input projection: h = x @ W_in + b_in
    gemm(x, wtIn, b_in, hA, NN, FF, DD, 0, 0);

    // 2) two GAT layers
    auto gat = [&](const float* hin, float* hout, const __bf16* Wt,
                   const float* as_, const float* ad_, const float* bg) {
        gemm(hin, Wt, nullptr, G, NN, DD, DD, 0, 0);
        gat_scores<<<(NN + 255) / 256, 256, 0, stream>>>(G, as_, ad_, sS, sD);
        gat_init<<<(NN * DD + 255) / 256, 256, 0, stream>>>(mBuf, dBuf, hout, bg);
        gat_edge1<<<(ETOT + 255) / 256, 256, 0, stream>>>(ei, sS, sD, eV, mBuf);
        gat_edge2<<<(ETOT + 255) / 256, 256, 0, stream>>>(ei, eV, mBuf, dBuf);
        gat_edge3<<<(ETOT + 7) / 8, 256, 0, stream>>>(ei, eV, dBuf, G, hout);
    };
    gat(hA, hB, wtG1, as1, ad1, bg1);
    gat(hB, hA, wtG2, as2, ad2, bg2);

    // 3) cluster gather + embedding, prepend cls token
    gather_nodes<<<(NN * DD + 255) / 256, 256, 0, stream>>>(hA, order, G);
    copy_cls<<<1, 128, 0, stream>>>(clstok, tBuf);
    gemm(G, wtEmb, b_emb, tBuf + DD, KC, CSZ * DD, DD, 0, 0);

    // 4) transformer, 8 pre-norm layers
    for (int l = 0; l < LL; ++l) {
        layernorm<<<(TT + 255) / 256, 256, 0, stream>>>(tBuf, ln1_s + l * DD, ln1_b + l * DD, uBuf, TT);
        gemm(uBuf, wtQkv + (size_t)l * DD * 3 * DD, bqkv + l * 3 * DD, qkvB, TT, DD, 3 * DD, 0, 0);
        attention<<<(TT * HH + 255) / 256, 256, 0, stream>>>(qkvB, atnB);
        gemm(atnB, wtO + (size_t)l * DD * DD, bo + l * DD, tBuf, TT, DD, DD, 0, 1);
        layernorm<<<(TT + 255) / 256, 256, 0, stream>>>(tBuf, ln2_s + l * DD, ln2_b + l * DD, uBuf, TT);
        gemm(uBuf, wtF1 + (size_t)l * DD * 4 * DD, bf1 + l * 4 * DD, midB, TT, DD, 4 * DD, 1, 0);
        gemm(midB, wtF2 + (size_t)l * 4 * DD * DD, bf2 + l * DD, tBuf, TT, 4 * DD, DD, 0, 1);
    }

    // 5) classifier head on cls token
    classifier<<<1, 32, 0, stream>>>(tBuf, Wc, bc, out);
}
